// SecurityAwareAttention_88021059764888
// MI455X (gfx1250) — matrix operand order repaired and checked by reading, not compiled, not run on hardware
//
#include <hip/hip_runtime.h>
#include <hip/hip_bf16.h>
#include <math.h>

// ---------------------------------------------------------------------------
// SecurityAwareAttention for gfx1250 (MI455X), wave32 + WMMA f16->f32.
//
// Math note: sec_bias = security @ Ws^T + bs is broadcast over the KEY axis
// and added before a softmax over that same axis -> per-row constant cancels
// exactly. probs and out are mathematically independent of the security
// branch, so it is skipped entirely.
//
// Precision: activations/weights converted to f16 once; all GEMMs use
// v_wmma_f32_16x16x32_f16 (f32 accumulate). Softmax and the probs output are
// full f32.
//
// Roofline: dominated by the 268MB probs tensor. We write it once with
// non-temporal stores (never re-read from HBM): P@V is fused into the softmax
// kernel and consumes the probabilities straight from LDS. Remaining traffic
// ~330MB => ~14us floor at 23.3 TB/s; 52 GFLOP of f16 WMMA is far below the
// matrix ceiling and hides under the memory streams.
// ---------------------------------------------------------------------------

#define HIDDEN   1024
#define HEADS    16
#define HEAD_DIM 64
#define BATCH    4
#define SEQ      1024
#define ROWS     (BATCH * SEQ)          // 4096

typedef __attribute__((ext_vector_type(16))) _Float16 v16h;
typedef __attribute__((ext_vector_type(8)))  _Float16 v8h;
typedef __attribute__((ext_vector_type(8)))  float    v8f;

// ---------------------------------------------------------------------------
// f32 -> f16 convert (grid-stride)
// ---------------------------------------------------------------------------
__global__ __launch_bounds__(256) void cvt_f32_to_f16(const float* __restrict__ src,
                                                      _Float16* __restrict__ dst,
                                                      int n) {
    for (int i = blockIdx.x * blockDim.x + threadIdx.x; i < n;
         i += gridDim.x * blockDim.x)
        dst[i] = (_Float16)src[i];
}

// ---------------------------------------------------------------------------
// WMMA fragment loaders.
// A 16x32 f16 (this file's contiguous k order, same as B): lane L = (g=L/16, r=L%16); row r of A-source;
//   elems 0..7  = k0 + g*16     .. +8
//   elems 8..15 = k0 + g*16 + 8 .. +8
// B 32x16 f16: column n = r of B-source (row-major over k); lane holds 16
//   contiguous k values starting at k0 + g*16.
// ---------------------------------------------------------------------------
__device__ inline v16h load_a_frag_f16(const _Float16* __restrict__ row, int k0, int g) {
    v8h a0 = *(const v8h*)(row + k0 + g * 16);
    v8h a1 = *(const v8h*)(row + k0 + g * 16 + 8);
    v16h a;
#pragma unroll
    for (int i = 0; i < 8; ++i) { a[i] = a0[i]; a[i + 8] = a1[i]; }
    return a;
}

__device__ inline v16h load_a_frag_f32(const float* row, int k0, int g) {
    v8f f0 = *(const v8f*)(row + k0 + g * 16);
    v8f f1 = *(const v8f*)(row + k0 + g * 16 + 8);
    v16h a;
#pragma unroll
    for (int i = 0; i < 8; ++i) { a[i] = (_Float16)f0[i]; a[i + 8] = (_Float16)f1[i]; }
    return a;
}

// ---------------------------------------------------------------------------
// QKV projection: dst = X @ W^T + bias.
// which: 0/1 -> store [b][h][n][d] (Q, K); 2 -> store transposed [b][h][d][n] (V)
// One 16x16 output tile per wave, K loop = 1024 in steps of 32 (32 WMMA).
// ---------------------------------------------------------------------------
__global__ __launch_bounds__(256) void qkv_gemm(const _Float16* __restrict__ Xh,
                                                const _Float16* __restrict__ Wh,
                                                const float* __restrict__ bias,
                                                _Float16* __restrict__ dst,
                                                int which) {
    const int wave = threadIdx.x >> 5, lane = threadIdx.x & 31;
    const int tile = blockIdx.x * 8 + wave;      // 0 .. 16383
    const int m0 = (tile >> 6) * 16;             // row tile (4096/16 = 256)
    const int n0 = (tile & 63) * 16;             // col tile (1024/16 = 64)
    const int r = lane & 15, g = lane >> 4;

    const _Float16* arow = Xh + (size_t)(m0 + r) * HIDDEN;
    const _Float16* brow = Wh + (size_t)(n0 + r) * HIDDEN;

    v8f c = {};
    for (int k0 = 0; k0 < HIDDEN; k0 += 32) {
        v16h a = load_a_frag_f16(arow, k0, g);
        v16h b = *(const v16h*)(brow + k0 + g * 16);
        c = __builtin_amdgcn_wmma_f32_16x16x32_f16(false, a, false, b,
                                                   (short)0, c, false, false);
    }

    const int col = n0 + r;                 // D: N = lane%16
    const float bv = bias[col];
    const int h = col >> 6, d = col & 63;
#pragma unroll
    for (int e = 0; e < 8; ++e) {
        const int m = m0 + e + 8 * g;       // D: M = e + 8*(lane/16)
        const int b = m >> 10, n = m & 1023;
        const _Float16 val = (_Float16)(c[e] + bv);
        if (which == 2)   // V stored transposed per head: [b][h][d][n]
            dst[(((size_t)(b * HEADS + h) * HEAD_DIM + d) * SEQ) + n] = val;
        else              // Q/K: [b][h][n][d]
            dst[(((size_t)(b * HEADS + h) * SEQ + n) * HEAD_DIM) + d] = val;
    }
}

// ---------------------------------------------------------------------------
// Fused attention: scores (WMMA) -> softmax -> probs (NT store to d_out)
//                  -> P@V (WMMA with A sourced from LDS) -> Oh.
// One block per (b, h, 16-query tile); 8 waves.
// Phase 1: 8 waves x 8 key-tiles x 2 WMMA -> 16x1024 f32 scores in LDS.
// Phase 2: per-row wave32-shuffle softmax; normalized probs written back to
//          LDS in place and streamed to global with non-temporal stores.
// Phase 3: P@V, split-K across waves (w0-3: keys 0-511 / w4-7: 512-1023),
//          A fragments read from LDS (f32 -> f16 cvt), B = V^T from L2.
//          Partials combined through the (dead) scores buffer.
// ---------------------------------------------------------------------------
__global__ __launch_bounds__(256) void attn_fused(const _Float16* __restrict__ Qh,
                                                  const _Float16* __restrict__ Kh,
                                                  const _Float16* __restrict__ Vth,
                                                  float* __restrict__ probs,
                                                  _Float16* __restrict__ Oh) {
    __shared__ __align__(16) float s_sc[16 * SEQ];   // 64 KB (of 320 KB WGP LDS)

    const int qt = blockIdx.x & 63;
    const int h  = (blockIdx.x >> 6) & 15;
    const int b  = blockIdx.x >> 10;
    const int bh = b * HEADS + h;
    const int q0 = qt * 16;

    const int wave = threadIdx.x >> 5, lane = threadIdx.x & 31;
    const int r = lane & 15, g = lane >> 4;

    // ---- Phase 1: scores = Q K^T -----------------------------------------
    const _Float16* Qb = Qh + (size_t)bh * SEQ * HEAD_DIM;
    const _Float16* Kb = Kh + (size_t)bh * SEQ * HEAD_DIM;
    const _Float16* qrow = Qb + (size_t)(q0 + r) * HEAD_DIM;

#pragma unroll
    for (int t = 0; t < 8; ++t) {
        const int n0 = (wave * 8 + t) * 16;             // key tile
        const _Float16* krow = Kb + (size_t)(n0 + r) * HEAD_DIM;
        v8f c = {};
#pragma unroll
        for (int d0 = 0; d0 < HEAD_DIM; d0 += 32) {
            v16h a  = load_a_frag_f16(qrow, d0, g);
            v16h bf = *(const v16h*)(krow + d0 + g * 16);
            c = __builtin_amdgcn_wmma_f32_16x16x32_f16(false, a, false, bf,
                                                       (short)0, c, false, false);
        }
#pragma unroll
        for (int e = 0; e < 8; ++e)
            s_sc[(e + 8 * g) * SEQ + n0 + r] = c[e];
    }
    __syncthreads();

    // ---- Phase 2: softmax (per-row additive bias cancels) ----------------
    const float inv_sqrt_d = 0.125f;   // 1/sqrt(64)
    for (int row = wave; row < 16; row += 8) {
        float* srow = s_sc + row * SEQ;
        float mx = -INFINITY;
        for (int k = lane; k < SEQ; k += 32) mx = fmaxf(mx, srow[k]);
#pragma unroll
        for (int off = 16; off; off >>= 1) mx = fmaxf(mx, __shfl_xor(mx, off, 32));

        float sum = 0.f;
        for (int k = lane; k < SEQ; k += 32) {
            const float e = __expf((srow[k] - mx) * inv_sqrt_d);
            srow[k] = e;
            sum += e;
        }
#pragma unroll
        for (int off = 16; off; off >>= 1) sum += __shfl_xor(sum, off, 32);

        const float inv = 1.0f / sum;
        float* prow = probs + (((size_t)bh * SEQ) + (q0 + row)) * SEQ;
        for (int k = lane; k < SEQ; k += 32) {
            const float p = srow[k] * inv;
            srow[k] = p;                               // normalized, reused in P@V
            __builtin_nontemporal_store(p, prow + k);  // streamed, keep L2 clean
        }
    }
    __syncthreads();

    // ---- Phase 3: out_h = P @ V (split-K over waves) ---------------------
    const int dt    = wave & 3;        // d-tile: 4 x 16 columns of head_dim
    const int khalf = wave >> 2;       // key range half
    const int kbase = khalf * 512;
    const _Float16* vrow = Vth + ((size_t)bh * HEAD_DIM + dt * 16 + r) * SEQ;
    const float* parow = s_sc + r * SEQ;               // probs row q0+r in LDS

    v8f c = {};
#pragma unroll
    for (int k0 = 0; k0 < 512; k0 += 32) {
        v16h a  = load_a_frag_f32(parow, kbase + k0, g);   // LDS -> f16 frag
        v16h bf = *(const v16h*)(vrow + kbase + k0 + g * 16);
        c = __builtin_amdgcn_wmma_f32_16x16x32_f16(false, a, false, bf,
                                                   (short)0, c, false, false);
    }

    // combine the two key-halves through the (now dead) scores buffer
    __syncthreads();                   // all LDS reads of s_sc done
    if (khalf == 1) {
#pragma unroll
        for (int e = 0; e < 8; ++e)
            s_sc[(dt * 8 + e) * 32 + lane] = c[e];
    }
    __syncthreads();
    if (khalf == 0) {
        const int d = dt * 16 + r;
#pragma unroll
        for (int e = 0; e < 8; ++e) {
            const float v = c[e] + s_sc[(dt * 8 + e) * 32 + lane];
            const int q = q0 + e + 8 * g;
            Oh[((size_t)(b * SEQ + q) * HIDDEN) + h * HEAD_DIM + d] = (_Float16)v;
        }
    }
}

// ---------------------------------------------------------------------------
// Output projection: out = O @ Wo^T + bo  (f32 result to d_out).
// ---------------------------------------------------------------------------
__global__ __launch_bounds__(256) void out_gemm(const _Float16* __restrict__ Oh,
                                                const _Float16* __restrict__ Woh,
                                                const float* __restrict__ bo,
                                                float* __restrict__ out) {
    const int wave = threadIdx.x >> 5, lane = threadIdx.x & 31;
    const int tile = blockIdx.x * 8 + wave;
    const int m0 = (tile >> 6) * 16;
    const int n0 = (tile & 63) * 16;
    const int r = lane & 15, g = lane >> 4;

    const _Float16* arow = Oh  + (size_t)(m0 + r) * HIDDEN;
    const _Float16* brow = Woh + (size_t)(n0 + r) * HIDDEN;

    v8f c = {};
    for (int k0 = 0; k0 < HIDDEN; k0 += 32) {
        v16h a = load_a_frag_f16(arow, k0, g);
        v16h b = *(const v16h*)(brow + k0 + g * 16);
        c = __builtin_amdgcn_wmma_f32_16x16x32_f16(false, a, false, b,
                                                   (short)0, c, false, false);
    }

    const int col = n0 + r;
    const float bv = bo[col];
#pragma unroll
    for (int e = 0; e < 8; ++e) {
        const int m = m0 + e + 8 * g;
        out[(size_t)m * HIDDEN + col] = c[e] + bv;
    }
}

// ---------------------------------------------------------------------------
// Host launch. Workspace layout (f16), total ~48 MB:
//   Xh 8MB | Wqh 2MB | Wkh 2MB | Wvh 2MB | Woh 2MB | Qh 8MB | Kh 8MB
//   | Vth 8MB | Oh 8MB
// d_out: out f32 (4M elems) then probs f32 (64M elems).
// ---------------------------------------------------------------------------
extern "C" void kernel_launch(void* const* d_in, const int* in_sizes, int n_in,
                              void* d_out, int out_size, void* d_ws, size_t ws_size,
                              hipStream_t stream) {
    const float* X  = (const float*)d_in[0];
    // d_in[1] security_features, d_in[8] Ws, d_in[9] bs: provably no-ops (bias
    // cancels in softmax over the key axis) -> unused.
    const float* Wq = (const float*)d_in[2];
    const float* bq = (const float*)d_in[3];
    const float* Wk = (const float*)d_in[4];
    const float* bk = (const float*)d_in[5];
    const float* Wv = (const float*)d_in[6];
    const float* bv = (const float*)d_in[7];
    const float* Wo = (const float*)d_in[10];
    const float* bo = (const float*)d_in[11];

    char* ws = (char*)d_ws;
    const size_t XH_BYTES = (size_t)ROWS * HIDDEN * 2;      // 8 MB
    const size_t W_BYTES  = (size_t)HIDDEN * HIDDEN * 2;    // 2 MB
    _Float16* Xh  = (_Float16*)(ws);
    _Float16* Wqh = (_Float16*)(ws + XH_BYTES);
    _Float16* Wkh = (_Float16*)(ws + XH_BYTES + 1 * W_BYTES);
    _Float16* Wvh = (_Float16*)(ws + XH_BYTES + 2 * W_BYTES);
    _Float16* Woh = (_Float16*)(ws + XH_BYTES + 3 * W_BYTES);
    _Float16* Qh  = (_Float16*)(ws + XH_BYTES + 4 * W_BYTES);
    _Float16* Kh  = (_Float16*)(ws + 2 * XH_BYTES + 4 * W_BYTES);
    _Float16* Vth = (_Float16*)(ws + 3 * XH_BYTES + 4 * W_BYTES);
    _Float16* Oh  = (_Float16*)(ws + 4 * XH_BYTES + 4 * W_BYTES);

    float* out_ptr   = (float*)d_out;
    float* probs_ptr = out_ptr + (size_t)ROWS * HIDDEN;     // after `out`

    // 1) f32 -> f16 converts
    cvt_f32_to_f16<<<2048, 256, 0, stream>>>(X,  Xh,  ROWS * HIDDEN);
    cvt_f32_to_f16<<<1024, 256, 0, stream>>>(Wq, Wqh, HIDDEN * HIDDEN);
    cvt_f32_to_f16<<<1024, 256, 0, stream>>>(Wk, Wkh, HIDDEN * HIDDEN);
    cvt_f32_to_f16<<<1024, 256, 0, stream>>>(Wv, Wvh, HIDDEN * HIDDEN);
    cvt_f32_to_f16<<<1024, 256, 0, stream>>>(Wo, Woh, HIDDEN * HIDDEN);

    // 2) QKV projections (16384 tiles / 8 waves = 2048 blocks each)
    qkv_gemm<<<2048, 256, 0, stream>>>(Xh, Wqh, bq, Qh,  0);
    qkv_gemm<<<2048, 256, 0, stream>>>(Xh, Wkh, bk, Kh,  1);
    qkv_gemm<<<2048, 256, 0, stream>>>(Xh, Wvh, bv, Vth, 2);

    // 3) fused scores + softmax + probs + P@V (B*H*N/16 = 4096 blocks)
    attn_fused<<<4096, 256, 0, stream>>>(Qh, Kh, Vth, probs_ptr, Oh);

    // 4) output projection -> d_out
    out_gemm<<<2048, 256, 0, stream>>>(Oh, Woh, bo, out_ptr);
}
